// AFMLayer_60352880443631
// MI455X (gfx1250) — compile-verified, matrix operand order checked by reading, not run on hardware
//
#include <hip/hip_runtime.h>

typedef __attribute__((ext_vector_type(16))) _Float16 v16h;
typedef __attribute__((ext_vector_type(8)))  float    v8f;
typedef __attribute__((ext_vector_type(4)))  float    v4f;

#define NF     32
#define NPAIR  496        // 32*31/2
#define EDIM   64
#define ADIM   64
#define EPAD   4          // row pad: keeps 16B alignment, stride 68 ≡ 4 mod 64 banks
#define MT     31         // 496 / 16 m-tiles
#define NT     4          // 64 / 16 n-tiles
#define KT     2          // 64 / 32 k-steps

// K index inside a 16-bit 16x32 A/B fragment for VGPR j (0..7), lane-half h (0/1):
// j<4 -> 2j + 8h ; j>=4 -> 16 + 2(j-4) + 8h
__device__ __forceinline__ int kmap(int j, int h) {
    return ((j & 4) << 2) + 2 * (j & 3) + 8 * h;
}

__global__ __launch_bounds__(256) void afm_kernel(
    const float* __restrict__ embeds,   // [B, 32, 64]
    const float* __restrict__ attn_W,   // [64, 64]
    const float* __restrict__ attn_b,   // [64]
    const float* __restrict__ proj_h,   // [64, 1]
    const float* __restrict__ proj_p,   // [64, 1]
    float* __restrict__ out)            // [B, 1]
{
    __shared__ __align__(16) float s_emb[NF][EDIM + EPAD];
    __shared__ _Float16  s_w[EDIM][ADIM + 2];   // f16 attn_W, k-major
    __shared__ float     s_bias[ADIM];
    __shared__ float     s_h[ADIM];
    __shared__ float     s_p[EDIM];
    __shared__ float     s_logit[NPAIR];
    __shared__ unsigned char s_pr[NPAIR], s_pc[NPAIR];
    __shared__ float     s_red[256];
    __shared__ float     s_acc[4][EDIM];

    const int tid  = threadIdx.x;
    const int b    = blockIdx.x;
    const int lane = tid & 31;
    const int wave = tid >> 5;
    const int hf   = (lane >> 4) & 1;   // lane half (0: lanes 0-15, 1: 16-31)
    const int l16  = lane & 15;

    // ---- stage inputs in LDS (128-bit moves) ----
    const v4f* eb4 = (const v4f*)(embeds + (size_t)b * NF * EDIM);
    for (int j = tid; j < NF * EDIM / 4; j += 256) {
        const int row = j >> 4;             // 16 float4 per 64-float row
        const int col = (j & 15) * 4;
        *(v4f*)&s_emb[row][col] = eb4[j];
    }
    for (int i = tid; i < EDIM * ADIM; i += 256)
        s_w[i >> 6][i & 63] = (_Float16)attn_W[i];
    if (tid < ADIM) { s_bias[tid] = attn_b[tid]; s_h[tid] = proj_h[tid]; }
    if (tid < EDIM) { s_p[tid] = proj_p[tid]; }
    // upper-triangle pair index tables (p -> (r,c), r<c)
    for (int p = tid; p < NPAIR; p += 256) {
        int r = 0, off = 0;
        while (p >= off + (NF - 1 - r)) { off += NF - 1 - r; ++r; }
        s_pr[p] = (unsigned char)r;
        s_pc[p] = (unsigned char)(r + 1 + (p - off));
    }
    __syncthreads();

    // ---- preload B fragments (attn_W tiles) into registers ----
    v16h bf[KT][NT];
    #pragma unroll
    for (int kt = 0; kt < KT; ++kt) {
        #pragma unroll
        for (int nt = 0; nt < NT; ++nt) {
            v16h f;
            const int n = nt * 16 + l16;
            #pragma unroll
            for (int j = 0; j < 8; ++j) {
                const int k = kt * 32 + kmap(j, hf);
                f[2 * j]     = s_w[k][n];
                f[2 * j + 1] = s_w[k + 1][n];
            }
            bf[kt][nt] = f;
        }
    }

    // ---- GEMM: att = relu(bi@W + b); logits = att@h  (att never stored) ----
    for (int mt = wave; mt < MT; mt += 8) {
        const int p = mt * 16 + l16;            // this lane's pair row
        const float* er = s_emb[s_pr[p]];
        const float* ec = s_emb[s_pc[p]];

        // bi A-fragments built on the fly: two contiguous 8-float runs per k-step
        v16h af[KT];
        #pragma unroll
        for (int kt = 0; kt < KT; ++kt) {
            const int k0 = kt * 32 + 8 * hf;    // run 1: K = k0..k0+7 -> f[0..7]
            const int k1 = k0 + 16;             // run 2: K = k1..k1+7 -> f[8..15]
            const v4f p0 = (*(const v4f*)(er + k0))     * (*(const v4f*)(ec + k0));
            const v4f p1 = (*(const v4f*)(er + k0 + 4)) * (*(const v4f*)(ec + k0 + 4));
            const v4f p2 = (*(const v4f*)(er + k1))     * (*(const v4f*)(ec + k1));
            const v4f p3 = (*(const v4f*)(er + k1 + 4)) * (*(const v4f*)(ec + k1 + 4));
            v16h f;
            #pragma unroll
            for (int q = 0; q < 4; ++q) {
                f[q]      = (_Float16)p0[q];
                f[4 + q]  = (_Float16)p1[q];
                f[8 + q]  = (_Float16)p2[q];
                f[12 + q] = (_Float16)p3[q];
            }
            af[kt] = f;
        }

        float lp[8];
        #pragma unroll
        for (int i = 0; i < 8; ++i) lp[i] = 0.f;

        #pragma unroll
        for (int nt = 0; nt < NT; ++nt) {
            v8f d = {};
            d = __builtin_amdgcn_wmma_f32_16x16x32_f16(false, af[0], false, bf[0][nt],
                                                       (short)0, d, false, false);
            d = __builtin_amdgcn_wmma_f32_16x16x32_f16(false, af[1], false, bf[1][nt],
                                                       (short)0, d, false, false);
            const int   acol = nt * 16 + l16;
            const float hv   = s_h[acol];
            const float bv   = s_bias[acol];
            #pragma unroll
            for (int i = 0; i < 8; ++i)
                lp[i] += fmaxf(d[i] + bv, 0.f) * hv;   // fold projection_h
        }
        // reduce over the 16 lanes that share a D row
        #pragma unroll
        for (int i = 0; i < 8; ++i) {
            float v = lp[i];
            v += __shfl_xor(v, 1, 16);
            v += __shfl_xor(v, 2, 16);
            v += __shfl_xor(v, 4, 16);
            v += __shfl_xor(v, 8, 16);
            lp[i] = v;
        }
        if (l16 == 0) {
            #pragma unroll
            for (int i = 0; i < 8; ++i)
                s_logit[mt * 16 + i + 8 * hf] = lp[i];
        }
    }
    __syncthreads();

    // ---- softmax over 496 logits ----
    float l0 = (tid < NPAIR)       ? s_logit[tid]       : -3.4e38f;
    float l1 = (tid + 256 < NPAIR) ? s_logit[tid + 256] : -3.4e38f;
    s_red[tid] = fmaxf(l0, l1);
    __syncthreads();
    for (int s = 128; s > 0; s >>= 1) {
        if (tid < s) s_red[tid] = fmaxf(s_red[tid], s_red[tid + s]);
        __syncthreads();
    }
    const float gmax = s_red[0];
    __syncthreads();
    const float e0 = (tid < NPAIR)       ? __expf(l0 - gmax) : 0.f;
    const float e1 = (tid + 256 < NPAIR) ? __expf(l1 - gmax) : 0.f;
    s_red[tid] = e0 + e1;
    __syncthreads();
    for (int s = 128; s > 0; s >>= 1) {
        if (tid < s) s_red[tid] += s_red[tid + s];
        __syncthreads();
    }
    const float inv = 1.f / s_red[0];
    __syncthreads();
    if (tid < NPAIR)       s_logit[tid]       = e0 * inv;
    if (tid + 256 < NPAIR) s_logit[tid + 256] = e1 * inv;
    __syncthreads();

    // ---- attention_output[e] = sum_p w[p]*emb[r][e]*emb[c][e]; then @ proj_p ----
    const int e = tid & 63;
    const int g = tid >> 6;                 // 4 groups x 124 pairs
    float acc = 0.f;
    for (int p = g * 124; p < (g + 1) * 124; ++p)
        acc += s_logit[p] * s_emb[s_pr[p]][e] * s_emb[s_pc[p]][e];
    s_acc[g][e] = acc;
    __syncthreads();
    if (tid < 64) {
        const float ao = s_acc[0][tid] + s_acc[1][tid] + s_acc[2][tid] + s_acc[3][tid];
        s_red[tid] = ao * s_p[tid];
    }
    __syncthreads();
    if (tid == 0) {
        float sum = 0.f;
        #pragma unroll
        for (int i = 0; i < 64; ++i) sum += s_red[i];
        out[b] = sum;
    }
}

extern "C" void kernel_launch(void* const* d_in, const int* in_sizes, int n_in,
                              void* d_out, int out_size, void* d_ws, size_t ws_size,
                              hipStream_t stream) {
    const float* embeds = (const float*)d_in[0];
    const float* attn_W = (const float*)d_in[1];
    const float* attn_b = (const float*)d_in[2];
    const float* proj_h = (const float*)d_in[3];
    const float* proj_p = (const float*)d_in[4];
    float* out = (float*)d_out;

    const int B = in_sizes[0] / (NF * EDIM);   // 4096
    afm_kernel<<<B, 256, 0, stream>>>(embeds, attn_W, attn_b, proj_h, proj_p, out);
}
